// GraphFeatureTokenizer_38465727103229
// MI455X (gfx1250) — compile-verified, matrix operand order checked by reading, not run on hardware
//
#include <hip/hip_runtime.h>
#include <hip/hip_bf16.h>

typedef __attribute__((ext_vector_type(8)))  float          v8f;
typedef __attribute__((ext_vector_type(16))) __bf16         v16bf;
typedef __attribute__((ext_vector_type(8)))  unsigned short v8us;
typedef __attribute__((ext_vector_type(16))) unsigned short v16us;
typedef int v4i __attribute__((vector_size(16)));   // matches builtin's V4i param

#define NGRAPH 64
#define NPG    128
#define EPG    1024
#define DFEAT  128
#define DP     64
#define HID    768
#define INDIM  320
#define NNODE  (NGRAPH * NPG)        // 8192
#define NEDGE  (NGRAPH * EPG)        // 65536
#define SEQ    (1 + NPG + EPG)       // 1153
#define MROWS  (NNODE + NEDGE)       // 73728

// ---- CDNA5 async global->LDS copy path (guarded; falls back to load+ds_store) ----
#if defined(__has_builtin)
# if __has_builtin(__builtin_amdgcn_global_load_async_to_lds_b128)
#  define HAVE_ASYNC_LDS 1
# endif
#endif
#ifndef HAVE_ASYNC_LDS
# define HAVE_ASYNC_LDS 0
#endif

// ---------------- helpers ----------------

__device__ __forceinline__ unsigned short f2bf(float f) {
    unsigned int u = __builtin_bit_cast(unsigned int, f);
    u += 0x7FFFu + ((u >> 16) & 1u);          // round-to-nearest-even
    return (unsigned short)(u >> 16);
}

// copy 8 bf16 (16 bytes) global -> LDS
__device__ __forceinline__ void cp8h(unsigned short* l, const unsigned short* g) {
#if HAVE_ASYNC_LDS
    const v4i* gv = (const v4i*)g;
    v4i* lv = (v4i*)l;
    __builtin_amdgcn_global_load_async_to_lds_b128(
        (__attribute__((address_space(1))) v4i*)gv,
        (__attribute__((address_space(3))) v4i*)lv, 0, 0);
#else
    *(v8us*)l = *(const v8us*)g;
#endif
}

__device__ __forceinline__ void stage_fence() {
#if HAVE_ASYNC_LDS
# if defined(__has_builtin) && __has_builtin(__builtin_amdgcn_s_wait_asynccnt)
    __builtin_amdgcn_s_wait_asynccnt(0);
# else
    asm volatile("s_wait_asynccnt 0x0" ::: "memory");
# endif
#endif
}

// Load one 16x32 bf16 WMMA operand fragment from LDS (row stride LDA halves).
// lane<16: rows M=lane, K halves {0..7, 16..23}; lane>=16: M=lane-16, K {8..15, 24..31}
__device__ __forceinline__ v16bf load_frag(const unsigned short* p) {
    v8us a = *(const v8us*)(p);
    v8us b = *(const v8us*)(p + 16);
    v16us w = __builtin_shufflevector(a, b, 0,1,2,3,4,5,6,7,8,9,10,11,12,13,14,15);
    return __builtin_bit_cast(v16bf, w);
}

// map compute row (0..MROWS-1) -> element offset inside tokens[B,SEQ,HID]
__device__ __forceinline__ long out_off(int row, int col) {
    int b, s;
    if (row < NNODE) { b = row >> 7; s = 1 + (row & 127); }
    else { int e = row - NNODE; b = e >> 10; s = 1 + NPG + (e & 1023); }
    return ((long)(b * SEQ + s)) * HID + col;
}

// ---------------- kernel 0: bulk f32 -> bf16 conversion ----------------

__global__ __launch_bounds__(256) void f32_to_bf16_kernel(const float* __restrict__ src,
                                                          unsigned short* __restrict__ dst,
                                                          long n) {
    long i = ((long)blockIdx.x * 256 + threadIdx.x) * 8;
    if (i + 8 <= n) {
        float4 a = *(const float4*)(src + i);
        float4 b = *(const float4*)(src + i + 4);
        v8us o;
        o[0] = f2bf(a.x); o[1] = f2bf(a.y); o[2] = f2bf(a.z); o[3] = f2bf(a.w);
        o[4] = f2bf(b.x); o[5] = f2bf(b.y); o[6] = f2bf(b.z); o[7] = f2bf(b.w);
        *(v8us*)(dst + i) = o;
    } else {
        for (; i < n; ++i) dst[i] = f2bf(src[i]);
    }
}

// ---------------- kernel 1: batched MGS QR (first 64 columns) -> bf16 P ----------------

__global__ __launch_bounds__(512) void qr_mgs_kernel(const float* __restrict__ G,
                                                     unsigned short* __restrict__ Pb) {
    __shared__ float col[DP][NPG + 1];   // padded stride 129: bank-conflict free
    const int b    = blockIdx.x;
    const int t    = threadIdx.x;
    const int wave = t >> 5;
    const int lane = t & 31;

    #pragma unroll
    for (int i = 0; i < 16; ++i) {
        int lin = t + 512 * i;               // 8192 elems
        int c = lin & 63, r = lin >> 6;
        col[c][r] = G[(size_t)b * NPG * NPG + (size_t)r * NPG + c];
    }
    __syncthreads();

    for (int j = 0; j < DP; ++j) {
        if (wave == 0) {                      // normalize column j (one wave)
            float v0 = col[j][lane],      v1 = col[j][lane + 32];
            float v2 = col[j][lane + 64], v3 = col[j][lane + 96];
            float p = v0 * v0 + v1 * v1 + v2 * v2 + v3 * v3;
            #pragma unroll
            for (int m = 16; m >= 1; m >>= 1) p += __shfl_xor(p, m, 32);
            float rinv = rsqrtf(p);
            col[j][lane]      = v0 * rinv; col[j][lane + 32] = v1 * rinv;
            col[j][lane + 64] = v2 * rinv; col[j][lane + 96] = v3 * rinv;
        }
        __syncthreads();
        for (int k = j + 1 + wave; k < DP; k += 16) {
            float u0 = col[j][lane],      u1 = col[j][lane + 32];
            float u2 = col[j][lane + 64], u3 = col[j][lane + 96];
            float w0 = col[k][lane],      w1 = col[k][lane + 32];
            float w2 = col[k][lane + 64], w3 = col[k][lane + 96];
            float d = u0 * w0 + u1 * w1 + u2 * w2 + u3 * w3;
            #pragma unroll
            for (int m = 16; m >= 1; m >>= 1) d += __shfl_xor(d, m, 32);
            col[k][lane]      = w0 - d * u0; col[k][lane + 32] = w1 - d * u1;
            col[k][lane + 64] = w2 - d * u2; col[k][lane + 96] = w3 - d * u3;
        }
        __syncthreads();
    }

    #pragma unroll
    for (int i = 0; i < 16; ++i) {
        int lin = t + 512 * i;
        int c = lin & 63, r = lin >> 6;
        Pb[((size_t)(b * NPG + r)) * DP + c] = f2bf(col[c][r]);
    }
}

// ---------------- kernel 2: fused concat + bf16 WMMA GEMM + scatter ----------------

#define BM   128
#define BN   256
#define BK   32
#define LDA  40          // 32 halves + 8 pad (80B row stride, conflict-free b128)

__global__ __launch_bounds__(256) void tokenizer_gemm_kernel(
        const unsigned short* __restrict__ xb,   // [NNODE, DFEAT] bf16
        const unsigned short* __restrict__ eab,  // [NEDGE, DFEAT] bf16
        const unsigned short* __restrict__ Pb,   // [NNODE, DP]    bf16
        const unsigned short* __restrict__ EVb,  // [DP]           bf16
        const unsigned short* __restrict__ EEb,  // [DP]           bf16
        const unsigned short* __restrict__ Wb,   // [HID, INDIM]   bf16
        const float* __restrict__ bias,          // [HID] f32
        const int*   __restrict__ ei,            // [2, NEDGE]
        float* __restrict__ out) {
    __shared__ unsigned short As[BM * LDA];   // 10.0 KB
    __shared__ unsigned short Bs[BN * LDA];   // 20.0 KB
    __shared__ int idxU[BM];
    __shared__ int idxV[BM];

    const int t    = threadIdx.x;
    const int lane = t & 31;
    const int wave = t >> 5;
    const int wm   = wave & 1;                 // M half (64 rows)
    const int wn   = wave >> 1;                // N quarter (64 cols)

    const int mb = blockIdx.x / 3;             // 576 row-blocks
    const int nb = blockIdx.x % 3;             // 3 col-blocks of 256
    const int rowBase = mb * BM;
    const bool isNode = (rowBase < NNODE);
    const int edgeBase = rowBase - NNODE;

    if (t < BM) {
        if (isNode) { idxU[t] = rowBase + t; idxV[t] = rowBase + t; }
        else {
            int e = edgeBase + t;
            idxU[t] = ei[e];
            idxV[t] = ei[NEDGE + e];
        }
    }

    v8f acc[4][4] = {};
    const int half = (lane < 16) ? 0 : 8;
    const int mrow = lane & 15;

    for (int ks = 0; ks < INDIM / BK; ++ks) {
        const int koff = ks * BK;
        __syncthreads();

        // ---- stage A tile [BM x 32] bf16: concat regions fold into source select ----
        if (koff < DFEAT) {
            const unsigned short* src = isNode ? (xb + (size_t)rowBase * DFEAT)
                                               : (eab + (size_t)edgeBase * DFEAT);
            #pragma unroll
            for (int i = 0; i < 2; ++i) {
                int lin = t + 256 * i;             // 512 chunks of 8 halves
                int r = lin >> 2, c8 = lin & 3;
                cp8h(&As[r * LDA + c8 * 8], src + (size_t)r * DFEAT + koff + c8 * 8);
            }
        } else if (koff < DFEAT + 2 * DP) {
            const int* idx = (koff < DFEAT + DP) ? idxU : idxV;
            const int po = (koff < DFEAT + DP) ? (koff - DFEAT) : (koff - DFEAT - DP);
            #pragma unroll
            for (int i = 0; i < 2; ++i) {
                int lin = t + 256 * i;
                int r = lin >> 2, c8 = lin & 3;
                cp8h(&As[r * LDA + c8 * 8], Pb + (size_t)idx[r] * DP + po + c8 * 8);
            }
        } else {
            const unsigned short* Ec = isNode ? EVb : EEb;
            const int po = koff - (DFEAT + 2 * DP);
            #pragma unroll
            for (int i = 0; i < 2; ++i) {
                int lin = t + 256 * i;
                int r = lin >> 2, c8 = lin & 3;
                cp8h(&As[r * LDA + c8 * 8], Ec + po + c8 * 8);
            }
        }

        // ---- stage B tile [BN x 32] bf16 from pre-converted weights ----
        #pragma unroll
        for (int i = 0; i < 4; ++i) {
            int lin = t + 256 * i;                 // 1024 chunks of 8 halves
            int n = lin >> 2, c8 = lin & 3;
            cp8h(&Bs[n * LDA + c8 * 8],
                 Wb + (size_t)(nb * BN + n) * INDIM + koff + c8 * 8);
        }
        stage_fence();
        __syncthreads();

        // ---- WMMA: 4x4 tiles of 16x16 per wave ----
        v16bf afr[4];
        #pragma unroll
        for (int mt = 0; mt < 4; ++mt) {
            int r = wm * 64 + mt * 16 + mrow;
            afr[mt] = load_frag(&As[r * LDA + half]);
        }
        #pragma unroll
        for (int nt = 0; nt < 4; ++nt) {
            int n = wn * 64 + nt * 16 + mrow;
            v16bf bfr = load_frag(&Bs[n * LDA + half]);
            #pragma unroll
            for (int mt = 0; mt < 4; ++mt) {
                acc[mt][nt] = __builtin_amdgcn_wmma_f32_16x16x32_bf16(
                    false, afr[mt], false, bfr, (short)0, acc[mt][nt], false, false);
            }
        }
    }

    // ---- epilogue: bias + streaming scatter into padded [B, SEQ, HID] ----
    #pragma unroll
    for (int mt = 0; mt < 4; ++mt) {
        #pragma unroll
        for (int nt = 0; nt < 4; ++nt) {
            int row0 = rowBase + wm * 64 + mt * 16 + ((lane < 16) ? 0 : 8);
            int coln = nb * BN + wn * 64 + nt * 16 + (lane & 15);
            float bv = bias[coln];
            #pragma unroll
            for (int j = 0; j < 8; ++j) {
                __builtin_nontemporal_store(acc[mt][nt][j] + bv,
                                            &out[out_off(row0 + j, coln)]);
            }
        }
    }
}

// ---------------- kernel 3: graph-token rows + attention-mask tail ----------------

__global__ __launch_bounds__(256) void fill_kernel(const float* __restrict__ gt,
                                                   float* __restrict__ out,
                                                   long tail, long tokElems) {
    long i = (long)blockIdx.x * 256 + threadIdx.x;
    if (i < (long)NGRAPH * HID) {
        int b = (int)(i / HID), h = (int)(i % HID);
        out[((long)b * SEQ) * HID + h] = gt[h];
    }
    long mi = i - (long)NGRAPH * HID;
    if (mi >= 0 && mi < tail) out[tokElems + mi] = 1.0f;
}

// ---------------- launch ----------------

extern "C" void kernel_launch(void* const* d_in, const int* in_sizes, int n_in,
                              void* d_out, int out_size, void* d_ws, size_t ws_size,
                              hipStream_t stream) {
    const float* x    = (const float*)d_in[0];
    const float* ea   = (const float*)d_in[1];
    const float* G    = (const float*)d_in[2];
    const float* EV   = (const float*)d_in[3];
    const float* EE   = (const float*)d_in[4];
    const float* W    = (const float*)d_in[5];
    const float* bias = (const float*)d_in[6];
    const float* gt   = (const float*)d_in[7];
    const int*   ei   = (const int*)d_in[8];
    float* out = (float*)d_out;

    // workspace layout (bf16 halves): P | x | edge_attr | W | E_V | E_E  (~19.5 MB)
    unsigned short* Pb  = (unsigned short*)d_ws;
    unsigned short* xb  = Pb  + (size_t)NNODE * DP;
    unsigned short* eab = xb  + (size_t)NNODE * DFEAT;
    unsigned short* Wb  = eab + (size_t)NEDGE * DFEAT;
    unsigned short* EVb = Wb  + (size_t)HID * INDIM;
    unsigned short* EEb = EVb + DP;

    qr_mgs_kernel<<<NGRAPH, 512, 0, stream>>>(G, Pb);

    auto conv = [&](const float* s, unsigned short* d, long n) {
        int blocks = (int)((n / 8 + 255) / 256) + 1;
        f32_to_bf16_kernel<<<blocks, 256, 0, stream>>>(s, d, n);
    };
    conv(x,  xb,  (long)NNODE * DFEAT);
    conv(ea, eab, (long)NEDGE * DFEAT);
    conv(W,  Wb,  (long)HID * INDIM);
    conv(EV, EVb, DP);
    conv(EE, EEb, DP);

    tokenizer_gemm_kernel<<<(MROWS / BM) * (HID / BN), 256, 0, stream>>>(
        xb, eab, Pb, EVb, EEb, Wb, bias, ei, out);

    long tokElems = (long)NGRAPH * SEQ * HID;
    long tail = (long)out_size - tokElems;
    if (tail < 0) tail = 0;
    long fillWork = (long)NGRAPH * HID + tail;
    int fillBlocks = (int)((fillWork + 255) / 256);
    fill_kernel<<<fillBlocks, 256, 0, stream>>>(gt, out, tail, tokElems);
}
